// HybridAttentionBlock_18829136625825
// MI455X (gfx1250) — matrix operand order repaired and checked by reading, not compiled, not run on hardware
//
#include <hip/hip_runtime.h>

typedef __attribute__((ext_vector_type(16))) __bf16 v16bf;
typedef __attribute__((ext_vector_type(8)))  __bf16 v8bf;
typedef __attribute__((ext_vector_type(8)))  float  v8f;

#define WIN 7
#define SHIFT 3
#define NH 12
#define DIM 384
#define HD 32
#define NTOK 49
#define IMG 56
#define NB 32
#define NWIN 64                 /* windows per image (8x8) */
#define BWIN (NB * NWIN)        /* 2048 windows total */
#define MROWS (BWIN * NTOK)     /* 100352 token rows */
#define SCALE_Q 0.17677669529663689f  /* 32^-0.5 */

__device__ __forceinline__ int regof(int h) { return h < 49 ? 0 : (h < 53 ? 1 : 2); }

/* ---------------- weight casts ---------------- */
__global__ void cast_qkvw_kernel(const float* __restrict__ src, __bf16* __restrict__ dst) {
  int idx = blockIdx.x * 256 + threadIdx.x;
  if (idx < DIM * 3 * DIM) {
    int n = idx % (3 * DIM);
    float v = src[idx];
    if (n < DIM) v *= SCALE_Q;          /* fold attention scale into Q weights */
    dst[idx] = (__bf16)v;
  }
}

__global__ void cast_kernel(const float* __restrict__ src, __bf16* __restrict__ dst, int count) {
  int idx = blockIdx.x * 256 + threadIdx.x;
  if (idx < count) dst[idx] = (__bf16)src[idx];
}

/* ---------- LN1 + cyclic shift + window partition -> bf16 ---------- */
__global__ __launch_bounds__(128)
void ln1_shift_window_kernel(const float* __restrict__ x, const float* __restrict__ g,
                             const float* __restrict__ bb, __bf16* __restrict__ xw) {
  __shared__ float s1[128], s2[128];
  const int tid = threadIdx.x;
  const size_t tok = blockIdx.x;
  const float* xp = x + tok * DIM;
  float v0 = xp[tid], v1 = xp[tid + 128], v2 = xp[tid + 256];
  s1[tid] = v0 + v1 + v2;
  s2[tid] = v0 * v0 + v1 * v1 + v2 * v2;
  __syncthreads();
  for (int off = 64; off > 0; off >>= 1) {
    if (tid < off) { s1[tid] += s1[tid + off]; s2[tid] += s2[tid + off]; }
    __syncthreads();
  }
  float mean = s1[0] * (1.0f / DIM);
  float var  = s2[0] * (1.0f / DIM) - mean * mean;
  float rstd = rsqrtf(var + 1e-5f);

  int b   = (int)(tok / (IMG * IMG));
  int pos = (int)(tok % (IMG * IMG));
  int hh = pos / IMG, ww = pos % IMG;
  int hs = hh - SHIFT; if (hs < 0) hs += IMG;   /* roll(-SHIFT): dest = (src - s) mod 56 */
  int ws2 = ww - SHIFT; if (ws2 < 0) ws2 += IMG;
  int dst = (b * NWIN + (hs / WIN) * 8 + (ws2 / WIN)) * NTOK + (hs % WIN) * WIN + (ws2 % WIN);
  __bf16* op = xw + (size_t)dst * DIM;
  op[tid]       = (__bf16)((v0 - mean) * rstd * g[tid] + bb[tid]);
  op[tid + 128] = (__bf16)((v1 - mean) * rstd * g[tid + 128] + bb[tid + 128]);
  op[tid + 256] = (__bf16)((v2 - mean) * rstd * g[tid + 256] + bb[tid + 256]);
}

/* ---------------- LN2 (row-aligned) ---------------- */
__global__ __launch_bounds__(128)
void ln2_kernel(const float* __restrict__ x2, const float* __restrict__ g,
                const float* __restrict__ bb, __bf16* __restrict__ h2) {
  __shared__ float s1[128], s2[128];
  const int tid = threadIdx.x;
  const size_t tok = blockIdx.x;
  const float* xp = x2 + tok * DIM;
  float v0 = xp[tid], v1 = xp[tid + 128], v2 = xp[tid + 256];
  s1[tid] = v0 + v1 + v2;
  s2[tid] = v0 * v0 + v1 * v1 + v2 * v2;
  __syncthreads();
  for (int off = 64; off > 0; off >>= 1) {
    if (tid < off) { s1[tid] += s1[tid + off]; s2[tid] += s2[tid + off]; }
    __syncthreads();
  }
  float mean = s1[0] * (1.0f / DIM);
  float var  = s2[0] * (1.0f / DIM) - mean * mean;
  float rstd = rsqrtf(var + 1e-5f);
  __bf16* op = h2 + tok * DIM;
  op[tid]       = (__bf16)((v0 - mean) * rstd * g[tid] + bb[tid]);
  op[tid + 128] = (__bf16)((v1 - mean) * rstd * g[tid + 128] + bb[tid + 128]);
  op[tid + 256] = (__bf16)((v2 - mean) * rstd * g[tid + 256] + bb[tid + 256]);
}

/* ------------- A-panel-resident bf16 GEMM: A read ONCE from HBM -------------
   Block = 256 threads (8 waves). A panel (BMT x KK) staged to LDS in the WMMA
   A-fragment layout; loop over N in 128-wide chunks with B restaged per K-step
   (weights are L2-resident). Wave w = one 16-wide N column, BMT/16 m-tiles.
   EPI 0: qkv  (out bf16, +bias, Q-bias scaled)
   EPI 1: proj (window-reverse + unshift + residual -> fp32 x2)
   EPI 2: fc1  (+bias, exact GELU -> bf16)
   EPI 3: fc2  (+bias, +x2 residual -> fp32 d_out)                              */
template <int EPI, int KK, int BMT>
__global__ __launch_bounds__(256)
void gemm_bf16(const __bf16* __restrict__ A, const __bf16* __restrict__ Wt,
               const float* __restrict__ bias, const float* __restrict__ extra,
               float* __restrict__ outf, __bf16* __restrict__ outb, int Nc) {
  constexpr int MT = BMT / 16;              /* m-tiles per block   */
  constexpr int KC = KK / 32;               /* 32-wide k-chunks    */
  __shared__ __bf16 Apanel[MT * KC * 512];  /* <= 48 KB            */
  __shared__ __bf16 Bf[8][512];             /* 8 KB                */
  const int tid  = threadIdx.x;
  const int lane = tid & 31;
  const int wn   = tid >> 5;                /* 0..7 n-tile column  */
  const int m0   = blockIdx.x * BMT;

  /* stage A panel: row-major global -> WMMA A-fragment layout */
  for (int p = tid; p < BMT * KK / 8; p += 256) {
    int f = p * 8;
    int m = f / KK, kg = f % KK;
    int kc = kg >> 5, kk = kg & 31;          /* kk in {0,8,16,24} */
    int lane_s = (m & 15) + (((kk >> 3) & 1) << 4);
    int eb = (kk < 16) ? 0 : 8;
    v8bf av = *(const v8bf*)(A + (size_t)(m0 + m) * KK + kg);
    *(v8bf*)(&Apanel[((m >> 4) * KC + kc) * 512 + lane_s * 16 + eb]) = av;
  }

  /* B staging indices: thread -> 16 contiguous N of one K row */
  const int bk = tid >> 3;                  /* 0..31 */
  const int bn = (tid & 7) << 4;            /* 0..112 */
  const int b_lanebase = ((bk >> 3) & 1) << 4;
  const int b_e = (bk & 7) + ((bk >> 4) << 3);
  const int rbase = (lane >> 4) << 3;       /* 0 or 8 */
  const int cl = lane & 15;

  for (int n0 = 0; n0 < Nc; n0 += 128) {
    v8f acc[MT];
    for (int mi = 0; mi < MT; mi++)
      for (int r = 0; r < 8; r++) acc[mi][r] = 0.0f;

    for (int k0 = 0; k0 < KK; k0 += 32) {
      if (k0 + 32 < KK)  /* gfx1250 global_prefetch_b8 on next B k-slice */
        __builtin_prefetch(Wt + (size_t)(k0 + 32 + bk) * Nc + n0 + bn, 0, 1);
      v16bf bval = *(const v16bf*)(Wt + (size_t)(k0 + bk) * Nc + n0 + bn);
      __syncthreads();
#pragma unroll
      for (int ii = 0; ii < 16; ii++)
        Bf[bn >> 4][(b_lanebase + ii) * 16 + b_e] = bval[ii];
      __syncthreads();
      v16bf bfrag = *(const v16bf*)(&Bf[wn][lane * 16]);
      const int kc = k0 >> 5;
#pragma unroll
      for (int mi = 0; mi < MT; mi++) {
        v16bf afrag = *(const v16bf*)(&Apanel[(mi * KC + kc) * 512 + lane * 16]);
        acc[mi] = __builtin_amdgcn_wmma_f32_16x16x32_bf16(
            false, afrag, false, bfrag, (short)0, acc[mi], false, false);
      }
    }

    /* fused epilogue for this N chunk */
    for (int mi = 0; mi < MT; mi++) {
#pragma unroll
      for (int r = 0; r < 8; r++) {
        int row = m0 + mi * 16 + r + rbase;
        int col = n0 + wn * 16 + cl;
        float v = acc[mi][r];
        if (EPI == 0) {
          float bsc = (col < DIM) ? SCALE_Q : 1.0f;
          outb[(size_t)row * Nc + col] = (__bf16)(v + bias[col] * bsc);
        } else if (EPI == 1) {
          int wi = row / NTOK, t = row % NTOK;
          int bb2 = wi >> 6, wrem = wi & 63;
          int wh = wrem >> 3, wwn = wrem & 7;
          int i = t / WIN, j = t % WIN;
          int hh = wh * WIN + i + SHIFT;  if (hh >= IMG) hh -= IMG;
          int w2 = wwn * WIN + j + SHIFT; if (w2 >= IMG) w2 -= IMG;
          size_t dst = ((size_t)bb2 * (IMG * IMG) + hh * IMG + w2) * DIM + col;
          outf[dst] = extra[dst] + v + bias[col];
        } else if (EPI == 2) {
          float t2 = v + bias[col];
          float ge = 0.5f * t2 * (1.0f + erff(t2 * 0.70710678118654752f));
          outb[(size_t)row * Nc + col] = (__bf16)ge;
        } else {
          size_t dst = (size_t)row * Nc + col;
          outf[dst] = extra[dst] + v + bias[col];
        }
      }
    }
  }
}

/* ---------------- windowed attention: one block per (window, head) ---------------- */
__global__ __launch_bounds__(128)
void attn_kernel(const __bf16* __restrict__ qkv, const float* __restrict__ bias_table,
                 __bf16* __restrict__ out) {
  __shared__ __bf16 Qf[4][512];       /* Q A-fragments, 4 m-tiles            */
  __shared__ __bf16 Kf[4][512];       /* K^T B-fragments, 4 n-tiles          */
  __shared__ __bf16 Vf[4][512];       /* V B-fragments [kc*2+nt]             */
  __shared__ float  Sm[64 * 64];      /* logits fp32                          */
  __shared__ __bf16 Pf[8][512];       /* softmax probs as A-fragments         */

  const int w = blockIdx.x;           /* 0..2047 */
  const int h = blockIdx.y;           /* 0..11   */
  const int tid = threadIdx.x;
  const int lane = tid & 31;
  const int wv = tid >> 5;

  const size_t rowbase = (size_t)w * NTOK * (3 * DIM);
  const int qoff = h * HD, koff = DIM + h * HD, voff = 2 * DIM + h * HD;

  for (int p = tid; p < 2048; p += 128) {
    int mt = p >> 9, ln = (p >> 4) & 31, e = p & 15;
    int tok = mt * 16 + (ln & 15);
    int half = ln >> 4;
    int d = (half << 4) + e;
    Qf[mt][ln * 16 + e] = (tok < NTOK) ? qkv[rowbase + (size_t)tok * (3 * DIM) + qoff + d] : (__bf16)0.0f;
  }
  for (int p = tid; p < 2048; p += 128) {
    int nt = p >> 9, ln = (p >> 4) & 31, e = p & 15;
    int tok = nt * 16 + (ln & 15);
    int d = ((ln >> 4) << 4) + e;
    Kf[nt][ln * 16 + e] = (tok < NTOK) ? qkv[rowbase + (size_t)tok * (3 * DIM) + koff + d] : (__bf16)0.0f;
  }
  for (int p = tid; p < 2048; p += 128) {
    int idx = p >> 9;                 /* kc*2 + nt */
    int kc = idx >> 1, nt = idx & 1;
    int ln = (p >> 4) & 31, e = p & 15;
    int tok = kc * 32 + ((e < 8) ? (8 * (ln >> 4) + e) : (16 + 8 * (ln >> 4) + (e - 8)));
    int d = nt * 16 + (ln & 15);
    Vf[idx][ln * 16 + e] = (tok < NTOK) ? qkv[rowbase + (size_t)tok * (3 * DIM) + voff + d] : (__bf16)0.0f;
  }
  __syncthreads();

  const int wrem = w & 63;
  const int wh = wrem >> 3, wwn = wrem & 7;

  /* S = Q K^T  (+ bias + shift mask), each wave one 16x64 row band */
  v16bf qfrag = *(const v16bf*)(&Qf[wv][lane * 16]);
#pragma unroll
  for (int nt = 0; nt < 4; nt++) {
    v16bf kfrag = *(const v16bf*)(&Kf[nt][lane * 16]);
    v8f s;
    for (int r = 0; r < 8; r++) s[r] = 0.0f;
    s = __builtin_amdgcn_wmma_f32_16x16x32_bf16(false, qfrag, false, kfrag, (short)0, s, false, false);
    int coltok = nt * 16 + (lane & 15);
#pragma unroll
    for (int r = 0; r < 8; r++) {
      int rowtok = wv * 16 + r + ((lane >> 4) << 3);
      float val = s[r];
      if (rowtok < NTOK && coltok < NTOK) {
        int in_ = rowtok / WIN, jn = rowtok % WIN;
        int im  = coltok / WIN, jm = coltok % WIN;
        int rel = (in_ - im + WIN - 1) * (2 * WIN - 1) + (jn - jm + WIN - 1);
        val += bias_table[rel * NH + h];
        int labn = regof(wh * WIN + in_) * 3 + regof(wwn * WIN + jn);
        int labm = regof(wh * WIN + im) * 3 + regof(wwn * WIN + jm);
        if (labn != labm) val -= 100.0f;
      }
      Sm[rowtok * 64 + coltok] = val;
    }
  }
  __syncthreads();

  /* fp32 softmax per row; scatter P into A-fragment layout (zero-padded) */
  if (tid < 64) {
    int row = tid;
    int mt = row >> 4, rl = row & 15;
    if (row < NTOK) {
      float mx = -1e30f;
      for (int c = 0; c < NTOK; c++) mx = fmaxf(mx, Sm[row * 64 + c]);
      float sum = 0.0f;
      for (int c = 0; c < NTOK; c++) sum += __expf(Sm[row * 64 + c] - mx);
      float inv = 1.0f / sum;
      for (int c = 0; c < 64; c++) {
        float pv = (c < NTOK) ? __expf(Sm[row * 64 + c] - mx) * inv : 0.0f;
        int kc = c >> 5, kk = c & 31;
        int half = (kk < 16) ? (kk >> 3) : ((kk - 16) >> 3);
        int e = (kk < 16) ? (kk & 7) : (8 + ((kk - 16) & 7));
        Pf[mt * 2 + kc][(rl + (half << 4)) * 16 + e] = (__bf16)pv;
      }
    } else {
      for (int c = 0; c < 64; c++) {
        int kc = c >> 5, kk = c & 31;
        int half = (kk < 16) ? (kk >> 3) : ((kk - 16) >> 3);
        int e = (kk < 16) ? (kk & 7) : (8 + ((kk - 16) & 7));
        Pf[mt * 2 + kc][(rl + (half << 4)) * 16 + e] = (__bf16)0.0f;
      }
    }
  }
  __syncthreads();

  /* O = P V, each wave one 16-row tile, K=64 split into 2 WMMAs */
#pragma unroll
  for (int nt = 0; nt < 2; nt++) {
    v8f o;
    for (int r = 0; r < 8; r++) o[r] = 0.0f;
#pragma unroll
    for (int kc = 0; kc < 2; kc++) {
      v16bf pf = *(const v16bf*)(&Pf[wv * 2 + kc][lane * 16]);
      v16bf vf = *(const v16bf*)(&Vf[kc * 2 + nt][lane * 16]);
      o = __builtin_amdgcn_wmma_f32_16x16x32_bf16(false, pf, false, vf, (short)0, o, false, false);
    }
    int col = nt * 16 + (lane & 15);
#pragma unroll
    for (int r = 0; r < 8; r++) {
      int row = wv * 16 + r + ((lane >> 4) << 3);
      if (row < NTOK)
        out[((size_t)w * NTOK + row) * DIM + h * HD + col] = (__bf16)o[r];
    }
  }
}

/* ---------------- launcher ---------------- */
extern "C" void kernel_launch(void* const* d_in, const int* in_sizes, int n_in,
                              void* d_out, int out_size, void* d_ws, size_t ws_size,
                              hipStream_t stream) {
  (void)in_sizes; (void)n_in; (void)out_size; (void)ws_size;
  const float* x        = (const float*)d_in[0];
  const float* norm1_g  = (const float*)d_in[1];
  const float* norm1_b  = (const float*)d_in[2];
  const float* qkv_w    = (const float*)d_in[3];
  const float* qkv_b    = (const float*)d_in[4];
  const float* proj_w   = (const float*)d_in[5];
  const float* proj_b   = (const float*)d_in[6];
  const float* btab     = (const float*)d_in[7];
  const float* norm2_g  = (const float*)d_in[8];
  const float* norm2_b  = (const float*)d_in[9];
  const float* fc1_w    = (const float*)d_in[10];
  const float* fc1_b    = (const float*)d_in[11];
  const float* fc2_w    = (const float*)d_in[12];
  const float* fc2_b    = (const float*)d_in[13];
  float* outp = (float*)d_out;

  char* ws = (char*)d_ws;
  const size_t SZ_QKVW  = (size_t)DIM * 3 * DIM * 2;
  const size_t SZ_PROJW = (size_t)DIM * DIM * 2;
  const size_t SZ_FC1W  = (size_t)DIM * 4 * DIM * 2;
  const size_t SZ_FC2W  = (size_t)4 * DIM * DIM * 2;
  const size_t SZ_XW    = (size_t)MROWS * DIM * 2;        /* bf16; reused as h2 */
  const size_t SZ_QKV   = (size_t)MROWS * 3 * DIM * 2;    /* bf16 */
  const size_t SZ_AOUT  = (size_t)MROWS * DIM * 2;        /* bf16; qkv+aout reused as m */
  const size_t SZ_X2    = (size_t)MROWS * DIM * 4;        /* fp32 */

  size_t off = 0;
  __bf16* w_qkv  = (__bf16*)(ws + off); off += SZ_QKVW;
  __bf16* w_proj = (__bf16*)(ws + off); off += SZ_PROJW;
  __bf16* w_fc1  = (__bf16*)(ws + off); off += SZ_FC1W;
  __bf16* w_fc2  = (__bf16*)(ws + off); off += SZ_FC2W;
  off = (off + 255) & ~(size_t)255;
  __bf16* xw     = (__bf16*)(ws + off);                    /* region A (later h2) */
  __bf16* h2     = xw;
  off += SZ_XW;
  __bf16* qkvbuf = (__bf16*)(ws + off);                    /* region B part 1 */
  __bf16* mbuf   = qkvbuf;                                 /* later: m (uses B1+B2) */
  off += SZ_QKV;
  __bf16* aout   = (__bf16*)(ws + off);                    /* region B part 2 */
  off += SZ_AOUT;
  float*  x2     = (float*)(ws + off);                     /* region C */
  off += SZ_X2;

  /* 1: weight casts */
  cast_qkvw_kernel<<<(DIM * 3 * DIM + 255) / 256, 256, 0, stream>>>(qkv_w, w_qkv);
  cast_kernel<<<(DIM * DIM + 255) / 256, 256, 0, stream>>>(proj_w, w_proj, DIM * DIM);
  cast_kernel<<<(DIM * 4 * DIM + 255) / 256, 256, 0, stream>>>(fc1_w, w_fc1, DIM * 4 * DIM);
  cast_kernel<<<(4 * DIM * DIM + 255) / 256, 256, 0, stream>>>(fc2_w, w_fc2, 4 * DIM * DIM);

  /* 2: LN1 + shift + window partition */
  ln1_shift_window_kernel<<<MROWS, 128, 0, stream>>>(x, norm1_g, norm1_b, xw);

  /* 3: QKV gemm  [100352,384] x [384,1152], A read once */
  gemm_bf16<0, 384, 64><<<MROWS / 64, 256, 0, stream>>>(
      xw, w_qkv, qkv_b, nullptr, nullptr, qkvbuf, 3 * DIM);

  /* 4: windowed attention */
  attn_kernel<<<dim3(BWIN, NH), 128, 0, stream>>>(qkvbuf, btab, aout);

  /* 5: proj gemm + window-reverse + unshift + residual -> x2 (fp32) */
  gemm_bf16<1, 384, 64><<<MROWS / 64, 256, 0, stream>>>(
      aout, w_proj, proj_b, x, x2, nullptr, DIM);

  /* 6: LN2 */
  ln2_kernel<<<MROWS, 128, 0, stream>>>(x2, norm2_g, norm2_b, h2);

  /* 7: fc1 + GELU  [100352,384] x [384,1536] */
  gemm_bf16<2, 384, 64><<<MROWS / 64, 256, 0, stream>>>(
      h2, w_fc1, fc1_b, nullptr, nullptr, mbuf, 4 * DIM);

  /* 8: fc2 + residual -> d_out  [100352,1536] x [1536,384], 16-row panels (48 KB LDS) */
  gemm_bf16<3, 1536, 16><<<MROWS / 16, 256, 0, stream>>>(
      mbuf, w_fc2, fc2_b, x2, outp, nullptr, DIM);
}
